// MambaMixer_22393959481412
// MI455X (gfx1250) — matrix operand order repaired and checked by reading, not compiled, not run on hardware
//
#include <hip/hip_runtime.h>
#include <hip/hip_bf16.h>

#define H_DIM 1024
#define I_DIM 2048
#define N_DIM 16
#define R_DIM 64
#define B_DIM 2
#define L_DIM 4096
#define M_DIM (B_DIM * L_DIM) // 8192 rows

typedef __attribute__((ext_vector_type(16))) __bf16 v16bf;
typedef __attribute__((ext_vector_type(8)))  __bf16 v8bf;
typedef __attribute__((ext_vector_type(8)))  float  v8f;

// ---------------------------------------------------------------------------
// Elementwise f32 -> (bf16 hi, bf16 lo residual) split, 4 elems/thread.
// ---------------------------------------------------------------------------
__global__ __launch_bounds__(256) void split_bf16_kernel(
    const float* __restrict__ src, __bf16* __restrict__ hi,
    __bf16* __restrict__ lo, int n4)
{
    int idx = blockIdx.x * 256 + threadIdx.x;
    if (idx >= n4) return;
    float4 v = ((const float4*)src)[idx];
    int b = idx * 4;
    __bf16 h0 = (__bf16)v.x, h1 = (__bf16)v.y, h2 = (__bf16)v.z, h3 = (__bf16)v.w;
    hi[b + 0] = h0; hi[b + 1] = h1; hi[b + 2] = h2; hi[b + 3] = h3;
    lo[b + 0] = (__bf16)(v.x - (float)h0);
    lo[b + 1] = (__bf16)(v.y - (float)h1);
    lo[b + 2] = (__bf16)(v.z - (float)h2);
    lo[b + 3] = (__bf16)(v.w - (float)h3);
}

// ---------------------------------------------------------------------------
// C(M x N) = A(M x K) * W(N x K)^T with pre-split bf16 hi/lo operands.
// bf16x3 split WMMA (hi*hi + hi*lo + lo*hi) -> fp32-grade accuracy at matrix
// rate.  One wave computes a (16*MT) x (16*NT) tile; every B hi/lo tile is
// reused MT times, every A hi/lo operand NT times -> inner loop is
// 4*MT + 4*NT loads : 3*MT*NT WMMAs per 32-K step.
// A operand (16x32 bf16): lane<16 holds K {k..k+7, k+16..k+23} of row lane%16;
// lane>=16 the +8-shifted chunks.  B operand (32x16): lane holds the same K chunks as A
// of column lane%16, chunk pair selected by lane group.
// ---------------------------------------------------------------------------
template<int MT, int NT>
__global__ __launch_bounds__(128) void gemm_wmma_bf16x3(
    const __bf16* __restrict__ Ahi, const __bf16* __restrict__ Alo,
    const __bf16* __restrict__ Whi, const __bf16* __restrict__ Wlo,
    float* __restrict__ C, int K, int ldc, int tiles_n)
{
    const int lane = threadIdx.x & 31;
    const int wid  = blockIdx.x * 4 + (threadIdx.x >> 5);
    const int mt   = wid / tiles_n;
    const int nb   = wid % tiles_n;
    const int l16  = lane & 15;
    const int half = lane >> 4;

    size_t abase[MT];
#pragma unroll
    for (int im = 0; im < MT; ++im)
        abase[im] = (size_t)(mt * (16 * MT) + im * 16 + l16) * K + half * 8;
    const int nbase = nb * (16 * NT);

    v8f zero = {};
    v8f acc[MT][NT];
#pragma unroll
    for (int im = 0; im < MT; ++im)
#pragma unroll
        for (int j = 0; j < NT; ++j) acc[im][j] = zero;

#pragma unroll 2
    for (int k = 0; k < K; k += 32) {
        v16bf ahi[MT], alo[MT];
#pragma unroll
        for (int im = 0; im < MT; ++im) {
            v8bf ah0 = *(const v8bf*)(Ahi + abase[im] + k);
            v8bf ah1 = *(const v8bf*)(Ahi + abase[im] + k + 16);
            v8bf al0 = *(const v8bf*)(Alo + abase[im] + k);
            v8bf al1 = *(const v8bf*)(Alo + abase[im] + k + 16);
            ahi[im] = __builtin_shufflevector(ah0, ah1, 0,1,2,3,4,5,6,7,8,9,10,11,12,13,14,15);
            alo[im] = __builtin_shufflevector(al0, al1, 0,1,2,3,4,5,6,7,8,9,10,11,12,13,14,15);
        }
#pragma unroll
        for (int j = 0; j < NT; ++j) {
            const size_t bofs = (size_t)(nbase + j * 16 + l16) * K + k + half * 8;
            v16bf bhi = __builtin_shufflevector(*(const v8bf*)(Whi + bofs), *(const v8bf*)(Whi + bofs + 16), 0,1,2,3,4,5,6,7,8,9,10,11,12,13,14,15);
            v16bf blo = __builtin_shufflevector(*(const v8bf*)(Wlo + bofs), *(const v8bf*)(Wlo + bofs + 16), 0,1,2,3,4,5,6,7,8,9,10,11,12,13,14,15);
#pragma unroll
            for (int im = 0; im < MT; ++im) {
                acc[im][j] = __builtin_amdgcn_wmma_f32_16x16x32_bf16(false, ahi[im], false, bhi, (short)0, acc[im][j], false, false);
                acc[im][j] = __builtin_amdgcn_wmma_f32_16x16x32_bf16(false, ahi[im], false, blo, (short)0, acc[im][j], false, false);
                acc[im][j] = __builtin_amdgcn_wmma_f32_16x16x32_bf16(false, alo[im], false, bhi, (short)0, acc[im][j], false, false);
            }
        }
    }
    // C/D layout: VGPR r -> row (r + half*8), col = lane%16
#pragma unroll
    for (int im = 0; im < MT; ++im)
#pragma unroll
        for (int j = 0; j < NT; ++j)
#pragma unroll
            for (int r = 0; r < 8; ++r) {
                C[(size_t)(mt * (16 * MT) + im * 16 + half * 8 + r) * ldc + nbase + j * 16 + l16] = acc[im][j][r];
            }
}

// ---------------------------------------------------------------------------
// Depthwise causal conv (K=4) + bias + SiLU; writes f32 xl (for the scan) and
// fused bf16 hi/lo (for the x_proj WMMA GEMM).
// ---------------------------------------------------------------------------
__global__ __launch_bounds__(256) void conv_silu_kernel(
    const float* __restrict__ xg, const float* __restrict__ cw,
    const float* __restrict__ cb, float* __restrict__ xl,
    __bf16* __restrict__ xlhi, __bf16* __restrict__ xllo)
{
    int idx = blockIdx.x * 256 + threadIdx.x;   // over M*I
    int i = idx & (I_DIM - 1);
    int m = idx >> 11;
    int t = m & (L_DIM - 1);
    float w0 = cw[i * 4 + 0], w1 = cw[i * 4 + 1], w2 = cw[i * 4 + 2], w3 = cw[i * 4 + 3];
    float acc = cb[i];
    acc += w3 * xg[(size_t)m * (2 * I_DIM) + i];                      // x[t]
    if (t >= 1) acc += w2 * xg[(size_t)(m - 1) * (2 * I_DIM) + i];    // x[t-1]
    if (t >= 2) acc += w1 * xg[(size_t)(m - 2) * (2 * I_DIM) + i];    // x[t-2]
    if (t >= 3) acc += w0 * xg[(size_t)(m - 3) * (2 * I_DIM) + i];    // x[t-3]
    float s = acc * (1.f / (1.f + __expf(-acc)));                     // SiLU
    size_t o = (size_t)m * I_DIM + i;
    xl[o] = s;
    __bf16 h = (__bf16)s;
    xlhi[o] = h;
    xllo[o] = (__bf16)(s - (float)h);
}

// ---------------------------------------------------------------------------
// dt = softplus(dt_low @ dt_proj_w^T + b).  K=64 -> exact f32 VALU (dt feeds
// exp() in the scan; exactness is cheap here). One block = one row m x 256 ch.
// ---------------------------------------------------------------------------
__global__ __launch_bounds__(256) void dt_softplus_kernel(
    const float* __restrict__ ssm, const float* __restrict__ dtw,
    const float* __restrict__ dtb, float* __restrict__ dtout)
{
    __shared__ float s[R_DIM];
    int m  = blockIdx.x >> 3;
    int ic = (blockIdx.x & 7) * 256 + threadIdx.x;
    if (threadIdx.x < R_DIM) s[threadIdx.x] = ssm[(size_t)m * 96 + threadIdx.x];
    __syncthreads();
    const float* w = dtw + (size_t)ic * R_DIM;
    float acc = dtb[ic];
#pragma unroll 8
    for (int r = 0; r < R_DIM; ++r) acc += s[r] * w[r];
    float sp = (acc > 20.f) ? acc : log1pf(expf(acc));
    dtout[(size_t)m * I_DIM + ic] = sp;
}

// ---------------------------------------------------------------------------
// Selective scan: thread per (b,i,n) state; 16-lane shfl_xor reduce over n.
// Software-pipelined (t+1 operands loaded ahead of the serial exp/FMA chain)
// + prefetch 8 steps ahead (working set is L2-resident).
// Writes y (incl. D*x skip) in place over xl.
// ---------------------------------------------------------------------------
__global__ __launch_bounds__(256) void scan_kernel(
    const float* __restrict__ dtbuf, const float* __restrict__ ssm,
    const float* __restrict__ Alog, const float* __restrict__ Dp,
    float* __restrict__ xl)
{
    int gid = blockIdx.x * 256 + threadIdx.x;   // over B*I*N = 65536
    int n  = gid & (N_DIM - 1);
    int ch = gid >> 4;                           // b*I + i
    int i  = ch & (I_DIM - 1);
    int b  = ch >> 11;
    float Aval = -expf(Alog[i * N_DIM + n]);
    float Dval = Dp[i];

    const float* xp = xl    + (size_t)b * L_DIM * I_DIM + i;
    const float* dp = dtbuf + (size_t)b * L_DIM * I_DIM + i;
    const float* bp = ssm   + (size_t)b * L_DIM * 96 + R_DIM + n;
    const float* cp = bp + N_DIM;
    float*       yp = xl    + (size_t)b * L_DIM * I_DIM + i;

    float x = xp[0], dt = dp[0], Bv = bp[0], Cv = cp[0];
    float h = 0.f;
    for (int t = 0; t < L_DIM; ++t) {
        int tn = (t + 1 < L_DIM) ? (t + 1) : t;
        float xn  = xp[(size_t)tn * I_DIM];
        float dtn = dp[(size_t)tn * I_DIM];
        float Bn  = bp[(size_t)tn * 96];
        float Cn  = cp[(size_t)tn * 96];
        if (t + 8 < L_DIM) {
            __builtin_prefetch(xp + (size_t)(t + 8) * I_DIM, 0, 0);
            __builtin_prefetch(dp + (size_t)(t + 8) * I_DIM, 0, 0);
        }
        h = h * __expf(dt * Aval) + (dt * Bv) * x;
        float p = h * Cv;
        p += __shfl_xor(p, 8, 32);
        p += __shfl_xor(p, 4, 32);
        p += __shfl_xor(p, 2, 32);
        p += __shfl_xor(p, 1, 32);
        if (n == 0) yp[(size_t)t * I_DIM] = p + Dval * x;   // all lanes read x first
        x = xn; dt = dtn; Bv = Bn; Cv = Cn;
    }
}

// ---------------------------------------------------------------------------
// y_gated = y * silu(gate), split straight to bf16 hi/lo for the out_proj GEMM.
// ---------------------------------------------------------------------------
__global__ __launch_bounds__(256) void gate_split_kernel(
    const float* __restrict__ xg, const float* __restrict__ y,
    __bf16* __restrict__ ghi, __bf16* __restrict__ glo)
{
    int idx = blockIdx.x * 256 + threadIdx.x;
    int i = idx & (I_DIM - 1);
    int m = idx >> 11;
    float g  = xg[(size_t)m * (2 * I_DIM) + I_DIM + i];
    float yv = y[(size_t)m * I_DIM + i];
    float v  = yv * (g * (1.f / (1.f + __expf(-g))));
    size_t o = (size_t)m * I_DIM + i;
    __bf16 h = (__bf16)v;
    ghi[o] = h;
    glo[o] = (__bf16)(v - (float)h);
}

extern "C" void kernel_launch(void* const* d_in, const int* in_sizes, int n_in,
                              void* d_out, int out_size, void* d_ws, size_t ws_size,
                              hipStream_t stream)
{
    const float* hidden  = (const float*)d_in[0];
    const float* in_w    = (const float*)d_in[1];
    const float* conv_w  = (const float*)d_in[2];
    const float* conv_b  = (const float*)d_in[3];
    const float* xproj_w = (const float*)d_in[4];
    const float* dt_w    = (const float*)d_in[5];
    const float* dt_b    = (const float*)d_in[6];
    const float* A_log   = (const float*)d_in[7];
    const float* Dp      = (const float*)d_in[8];
    const float* out_w   = (const float*)d_in[9];
    float* out = (float*)d_out;

    // ---- workspace layout -------------------------------------------------
    float* xg  = (float*)d_ws;                       // (M, 2I) proj        128 MB
    float* xl  = xg  + (size_t)M_DIM * 2 * I_DIM;    // (M, I) conv -> y     64 MB
    float* ssm = xl  + (size_t)M_DIM * I_DIM;        // (M, 96)               3 MB
    float* dtb = ssm + (size_t)M_DIM * 96;           // (M, I) dt            64 MB
    __bf16* hidhi = (__bf16*)(dtb + (size_t)M_DIM * I_DIM);
    __bf16* hidlo = hidhi + (size_t)M_DIM * H_DIM;            // hidden bf16  32 MB
    __bf16* w1hi  = hidlo + (size_t)M_DIM * H_DIM;
    __bf16* w1lo  = w1hi  + (size_t)2 * I_DIM * H_DIM;        // in_proj_w    16 MB
    __bf16* w3hi  = w1lo  + (size_t)2 * I_DIM * H_DIM;
    __bf16* w3lo  = w3hi  + (size_t)96 * I_DIM;               // x_proj_w    .75 MB
    __bf16* w7hi  = w3lo  + (size_t)96 * I_DIM;
    __bf16* w7lo  = w7hi  + (size_t)H_DIM * I_DIM;            // out_proj_w    8 MB
    __bf16* xlhi  = w7lo  + (size_t)H_DIM * I_DIM;
    __bf16* xllo  = xlhi  + (size_t)M_DIM * I_DIM;            // xl / y_gated 64 MB

    // ---- 0) one-time bf16 hi/lo splits ------------------------------------
    {
        int n4;
        n4 = (M_DIM * H_DIM) / 4;
        split_bf16_kernel<<<(n4 + 255) / 256, 256, 0, stream>>>(hidden, hidhi, hidlo, n4);
        n4 = (2 * I_DIM * H_DIM) / 4;
        split_bf16_kernel<<<(n4 + 255) / 256, 256, 0, stream>>>(in_w, w1hi, w1lo, n4);
        n4 = (96 * I_DIM) / 4;
        split_bf16_kernel<<<(n4 + 255) / 256, 256, 0, stream>>>(xproj_w, w3hi, w3lo, n4);
        n4 = (H_DIM * I_DIM) / 4;
        split_bf16_kernel<<<(n4 + 255) / 256, 256, 0, stream>>>(out_w, w7hi, w7lo, n4);
    }
    // ---- 1) proj = hidden @ in_proj_w^T : (8192 x 4096), K=1024 -----------
    {
        int tiles_n = (2 * I_DIM) / 64;                  // 64
        int waves   = (M_DIM / 32) * tiles_n;            // 16384 (32-row strips)
        gemm_wmma_bf16x3<2, 4><<<waves / 4, 128, 0, stream>>>(hidhi, hidlo, w1hi, w1lo, xg, H_DIM, 2 * I_DIM, tiles_n);
    }
    // ---- 2) depthwise conv + SiLU (+ fused bf16 split) --------------------
    conv_silu_kernel<<<(M_DIM * I_DIM) / 256, 256, 0, stream>>>(xg, conv_w, conv_b, xl, xlhi, xllo);
    // ---- 3) ssm = xl @ x_proj_w^T : (8192 x 96), K=2048 -------------------
    {
        int tiles_n = 96 / 48;                           // 2 (NT=3)
        int waves   = (M_DIM / 32) * tiles_n;            // 512
        gemm_wmma_bf16x3<2, 3><<<waves / 4, 128, 0, stream>>>(xlhi, xllo, w3hi, w3lo, ssm, I_DIM, 96, tiles_n);
    }
    // ---- 4) dt = softplus(dt_low @ dt_proj_w^T + b) -----------------------
    dt_softplus_kernel<<<M_DIM * (I_DIM / 256), 256, 0, stream>>>(ssm, dt_w, dt_b, dtb);
    // ---- 5) selective scan (y in place over xl) ---------------------------
    scan_kernel<<<(B_DIM * I_DIM * N_DIM) / 256, 256, 0, stream>>>(dtb, ssm, A_log, Dp, xl);
    // ---- 6) gating -> bf16 hi/lo (reuses xl bf16 buffers) -----------------
    gate_split_kernel<<<(M_DIM * I_DIM) / 256, 256, 0, stream>>>(xg, xl, xlhi, xllo);
    // ---- 7) out = y_gated @ out_proj_w^T : (8192 x 1024), K=2048 ----------
    {
        int tiles_n = H_DIM / 64;                        // 16
        int waves   = (M_DIM / 32) * tiles_n;            // 4096
        gemm_wmma_bf16x3<2, 4><<<waves / 4, 128, 0, stream>>>(xlhi, xllo, w7hi, w7lo, out, I_DIM, H_DIM, tiles_n);
    }
}
